// AttentionWithRelativeKey_2834678415814
// MI455X (gfx1250) — compile-verified, matrix operand order checked by reading, not run on hardware
//
#include <hip/hip_runtime.h>
#include <hip/hip_bf16.h>

// ---------------------------------------------------------------------------
// AttentionWithRelativeKey, MI455X (gfx1250, wave32, WMMA bf16)
// B=2, L=256, H=1024, NH=16, HD=64
// ---------------------------------------------------------------------------

#define B_   2
#define L_   256
#define H_   1024
#define NH_  16
#define HD_  64
#define ROWS 512      // B*L
#define H3_  3072

typedef __bf16 bf16;
typedef __attribute__((ext_vector_type(16))) __bf16 bf16x16;
typedef __attribute__((ext_vector_type(8)))  __bf16 bf16x8;
typedef __attribute__((ext_vector_type(8)))  float  f32x8;
typedef __attribute__((ext_vector_type(4)))  float  f32x4;

// ---- WMMA fragment loaders (ISA 7.12.2 16-bit A layout, wave32) ------------
// lane half 0: elements {0..7, 16..23}; lane half 1: elements {8..15, 24..31}
__device__ inline bf16x16 frag_from_bf16(const bf16* p, int half) {
    bf16x8 lo = *reinterpret_cast<const bf16x8*>(p + half * 8);
    bf16x8 hi = *reinterpret_cast<const bf16x8*>(p + 16 + half * 8);
    bf16x16 r;
#pragma unroll
    for (int i = 0; i < 8; ++i) { r[i] = lo[i]; r[8 + i] = hi[i]; }
    return r;
}

__device__ inline bf16x16 frag_from_f32(const float* p, int half) {
    f32x4 a = *reinterpret_cast<const f32x4*>(p + half * 8);
    f32x4 b = *reinterpret_cast<const f32x4*>(p + half * 8 + 4);
    f32x4 c = *reinterpret_cast<const f32x4*>(p + 16 + half * 8);
    f32x4 d = *reinterpret_cast<const f32x4*>(p + 16 + half * 8 + 4);
    bf16x16 r;
#pragma unroll
    for (int i = 0; i < 4; ++i) {
        r[i]      = (bf16)a[i];
        r[4 + i]  = (bf16)b[i];
        r[8 + i]  = (bf16)c[i];
        r[12 + i] = (bf16)d[i];
    }
    return r;
}

// Non-temporal variant for the single-use 512MB pos_emb stream.
__device__ inline bf16x16 frag_from_f32_nt(const float* p, int half) {
    f32x4 a = __builtin_nontemporal_load(reinterpret_cast<const f32x4*>(p + half * 8));
    f32x4 b = __builtin_nontemporal_load(reinterpret_cast<const f32x4*>(p + half * 8 + 4));
    f32x4 c = __builtin_nontemporal_load(reinterpret_cast<const f32x4*>(p + 16 + half * 8));
    f32x4 d = __builtin_nontemporal_load(reinterpret_cast<const f32x4*>(p + 16 + half * 8 + 4));
    bf16x16 r;
#pragma unroll
    for (int i = 0; i < 4; ++i) {
        r[i]      = (bf16)a[i];
        r[4 + i]  = (bf16)b[i];
        r[8 + i]  = (bf16)c[i];
        r[12 + i] = (bf16)d[i];
    }
    return r;
}

__device__ inline f32x8 wmma_bf16(bf16x16 a, bf16x16 b, f32x8 c) {
    return __builtin_amdgcn_wmma_f32_16x16x32_bf16(
        false, a, false, b, (short)0, c, false, false);
}

// ---------------------------------------------------------------------------
// Generic batched WMMA GEMM:  C[m,n] = sum_k A[m,k] * Bt[n,k] (+bias[n])
// 256 threads = 8 waves; each wave one 16x16 tile; WG tile 16(M) x 128(N).
// Batch z decomposed as (z/zdiv, z%zdiv) with separate strides per operand.
// ---------------------------------------------------------------------------
template <bool A_F32, bool B_F32, bool OUT_BF16, bool LEAKY>
__global__ __launch_bounds__(256) void wmma_gemm(
    const void* __restrict__ Ap, int lda, long long sA0, long long sA1,
    const void* __restrict__ Bp, int ldb, long long sB0, long long sB1,
    const float* __restrict__ bias,
    void* __restrict__ Cp, int ldc, long long sC0, long long sC1,
    int M, int N, int K, int zdiv)
{
    int z = blockIdx.z;
    long long offA = (long long)(z / zdiv) * sA0 + (long long)(z % zdiv) * sA1;
    long long offB = (long long)(z / zdiv) * sB0 + (long long)(z % zdiv) * sB1;
    long long offC = (long long)(z / zdiv) * sC0 + (long long)(z % zdiv) * sC1;

    int wid = threadIdx.x >> 5;
    int lane = threadIdx.x & 31;
    int col = lane & 15;
    int half = lane >> 4;

    int n0 = blockIdx.x * 128 + wid * 16;
    if (n0 >= N) return;
    int m0 = blockIdx.y * 16;
    (void)M;

    const float* Af = (const float*)Ap + offA;
    const bf16*  Ab = (const bf16*)Ap + offA;
    const float* Bf = (const float*)Bp + offB;
    const bf16*  Bb = (const bf16*)Bp + offB;

    f32x8 acc;
#pragma unroll
    for (int i = 0; i < 8; ++i) acc[i] = 0.0f;

    long long arow = (long long)(m0 + col) * lda;
    long long brow = (long long)(n0 + col) * ldb;

    for (int k0 = 0; k0 < K; k0 += 32) {
        bf16x16 a = A_F32 ? frag_from_f32(Af + arow + k0, half)
                          : frag_from_bf16(Ab + arow + k0, half);
        bf16x16 b = B_F32 ? frag_from_f32(Bf + brow + k0, half)
                          : frag_from_bf16(Bb + brow + k0, half);
        acc = wmma_bf16(a, b, acc);
    }

    float bv = bias ? bias[n0 + col] : 0.0f;
    float* Cf = (float*)Cp + offC;
    bf16*  Cb = (bf16*)Cp + offC;
#pragma unroll
    for (int r = 0; r < 8; ++r) {
        int row = m0 + r + half * 8;     // C/D layout: VGPR r -> M = r (+8 for hi half)
        float v = acc[r] + bv;
        if (LEAKY) v = v > 0.0f ? v : 0.01f * v;
        long long ci = (long long)row * ldc + n0 + col;
        if (OUT_BF16) Cb[ci] = (bf16)v;
        else          Cf[ci] = v;
    }
}

// ---------------------------------------------------------------------------
// WrT[h][e][d] = Wr[h*64+d][e]  (bf16) so the w-GEMM reads Bt rows contiguous
// ---------------------------------------------------------------------------
__global__ __launch_bounds__(256) void wr_transpose_kernel(
    const float* __restrict__ Wr, bf16* __restrict__ WrT)
{
    int i = blockIdx.x * 256 + threadIdx.x;      // NH*H*HD = 1048576
    int d = i & 63;
    int e = (i >> 6) & 1023;
    int h = i >> 16;
    WrT[i] = (bf16)Wr[(size_t)(h * 64 + d) * H_ + e];
}

// ---------------------------------------------------------------------------
// Split heads + add u/vb biases, build bf16 operands (incl. transposed V)
// ---------------------------------------------------------------------------
__global__ __launch_bounds__(256) void prep_heads_kernel(
    const float* __restrict__ qf, const float* __restrict__ kf,
    const float* __restrict__ vf, const float* __restrict__ u,
    const float* __restrict__ vb,
    bf16* __restrict__ qu, bf16* __restrict__ qv,
    bf16* __restrict__ kb, bf16* __restrict__ vT)
{
    int i = blockIdx.x * 256 + threadIdx.x;      // B*NH*L*HD = 524288
    int d = i & 63;
    int l = (i >> 6) & 255;
    int h = (i >> 14) & 15;
    int b = i >> 18;
    size_t src = ((size_t)(b * L_ + l)) * H_ + h * 64 + d;
    float q = qf[src];
    qu[i] = (bf16)(q + u[h * 64 + d]);
    qv[i] = (bf16)(q + vb[h * 64 + d]);
    kb[i] = (bf16)kf[src];
    size_t ti = (((size_t)(b * NH_ + h)) * 64 + d) * L_ + l;   // (b,h,d,l)
    vT[ti] = (bf16)vf[src];
}

// posc[b,h,q] = sum_d (q[b,h,q,d]+vb[h,d]) * br[h*64+d]   (rk bias fold)
__global__ __launch_bounds__(256) void posc_kernel(
    const float* __restrict__ qf, const float* __restrict__ vb,
    const float* __restrict__ br, float* __restrict__ posc)
{
    int i = blockIdx.x * 256 + threadIdx.x;      // B*NH*L = 8192
    int l = i & 255;
    int h = (i >> 8) & 15;
    int b = i >> 12;
    const float* qrow = qf + ((size_t)(b * L_ + l)) * H_ + h * 64;
    float s = 0.0f;
#pragma unroll
    for (int d = 0; d < 64; ++d) s += (qrow[d] + vb[h * 64 + d]) * br[h * 64 + d];
    posc[i] = s;
}

// ---------------------------------------------------------------------------
// The HBM-bound kernel: one WG per (b,q). Streams pos_emb[b,q] (1 MB) once,
// position = w(16x1024, LDS) x pos_emb^T via WMMA, + content + mask, softmax.
// The 32KB w tile is staged with CDNA5 GLOBAL_LOAD_ASYNC_TO_LDS_B128
// (ASYNCcnt-tracked DMA, no VGPR round-trip), then s_wait_asynccnt.
// ---------------------------------------------------------------------------
__global__ __launch_bounds__(256) void pos_softmax_kernel(
    const float* __restrict__ pos_emb, const bf16* __restrict__ w_b,
    const float* __restrict__ content, const float* __restrict__ posc,
    const unsigned char* __restrict__ mask, bf16* __restrict__ attn)
{
    __shared__ bf16  wlds[NH_ * H_];     // 32 KB: w[b,q,:, :]
    __shared__ float sc[NH_ * L_];       // 16 KB: scores
    __shared__ float red[NH_ * 16];
    __shared__ float rinv[NH_];

    int q = blockIdx.x;
    int b = blockIdx.y;
    int tid = threadIdx.x;

    {   // async global->LDS stage of the 16x1024 bf16 w tile (2048 x b128)
        const uint4* src = reinterpret_cast<const uint4*>(
            w_b + ((size_t)(b * L_ + q)) * (NH_ * H_));
        uint4* dst = reinterpret_cast<uint4*>(wlds);
#pragma unroll
        for (int j = 0; j < (NH_ * H_) / 8 / 256; ++j) {
            int i = tid + j * 256;
            unsigned lds_addr = (unsigned)(uintptr_t)(dst + i);   // low 32b = LDS offset
            unsigned long long gaddr = (unsigned long long)(uintptr_t)(src + i);
            asm volatile("global_load_async_to_lds_b128 %0, %1, off"
                         :: "v"(lds_addr), "v"(gaddr) : "memory");
        }
        asm volatile("s_wait_asynccnt 0x0" ::: "memory");
    }
    __syncthreads();

    int wid = tid >> 5, lane = tid & 31, col = lane & 15, half = lane >> 4;
    const float* pb = pos_emb + ((size_t)(b * L_ + q)) * L_ * H_;  // [key][e]
    const bf16* arow = wlds + col * H_;                            // A row = head
    int n0a = wid * 16;            // keys   0..127 strip
    int n0b = 128 + wid * 16;      // keys 128..255 strip

    f32x8 acc0, acc1;
#pragma unroll
    for (int i = 0; i < 8; ++i) { acc0[i] = 0.0f; acc1[i] = 0.0f; }

    const float* pr0 = pb + (size_t)(n0a + col) * H_;
    const float* pr1 = pb + (size_t)(n0b + col) * H_;
    for (int e0 = 0; e0 < H_; e0 += 32) {
        __builtin_prefetch(pr0 + e0 + 128, 0, 0);
        __builtin_prefetch(pr1 + e0 + 128, 0, 0);
        bf16x16 a  = frag_from_bf16(arow + e0, half);
        bf16x16 b0 = frag_from_f32_nt(pr0 + e0, half);
        acc0 = wmma_bf16(a, b0, acc0);
        bf16x16 b1 = frag_from_f32_nt(pr1 + e0, half);
        acc1 = wmma_bf16(a, b1, acc1);
    }

    // combine with content scores + rk-bias fold + mask -> LDS
#pragma unroll
    for (int t = 0; t < 2; ++t) {
        int n0 = t ? n0b : n0a;
        f32x8 ac = t ? acc1 : acc0;
        int key = n0 + col;
        bool ok = mask[b * L_ + key] != 0;
#pragma unroll
        for (int r = 0; r < 8; ++r) {
            int h = r + 8 * half;
            size_t bh = (size_t)(b * NH_ + h);
            float s = ac[r]
                    + content[(bh * L_ + q) * L_ + key]
                    + posc[bh * L_ + q];
            if (!ok) s = -1.0e6f;
            sc[h * L_ + key] = s;
        }
    }
    __syncthreads();

    // rowwise softmax over 256 keys; 16 threads per head row
    int r16 = tid >> 4, c16 = tid & 15;
    float mx = -3.0e38f;
    for (int j = c16; j < L_; j += 16) mx = fmaxf(mx, sc[r16 * L_ + j]);
    red[r16 * 16 + c16] = mx;
    __syncthreads();
    if (c16 == 0) {
        float m = red[r16 * 16];
#pragma unroll
        for (int j = 1; j < 16; ++j) m = fmaxf(m, red[r16 * 16 + j]);
        red[r16 * 16] = m;
    }
    __syncthreads();
    float rowm = red[r16 * 16];
    __syncthreads();
    float sum = 0.0f;
    for (int j = c16; j < L_; j += 16) {
        float e = __expf(sc[r16 * L_ + j] - rowm);
        sc[r16 * L_ + j] = e;
        sum += e;
    }
    red[r16 * 16 + c16] = sum;
    __syncthreads();
    if (c16 == 0) {
        float s = 0.0f;
#pragma unroll
        for (int j = 0; j < 16; ++j) s += red[r16 * 16 + j];
        rinv[r16] = 1.0f / s;
    }
    __syncthreads();

    // write attn probs bf16, layout (b,h,q,k), coalesced
    for (int i = tid; i < NH_ * L_; i += 256) {
        int h = i >> 8, k = i & 255;
        attn[(((size_t)(b * NH_ + h)) * L_ + q) * L_ + k] = (bf16)(sc[i] * rinv[h]);
    }
}

// ---------------------------------------------------------------------------
// Rowwise: y = LN( (leaky?) t + res ) * g + b ; optional f32 + bf16 outputs
// ---------------------------------------------------------------------------
__global__ __launch_bounds__(256) void ln_res_kernel(
    const float* __restrict__ t, const float* __restrict__ res,
    const float* __restrict__ g, const float* __restrict__ be,
    int leaky, float* __restrict__ outf, bf16* __restrict__ outb)
{
    __shared__ float red[256];
    int row = blockIdx.x, tid = threadIdx.x;
    const float* tr = t + (size_t)row * H_;
    const float* rr = res + (size_t)row * H_;

    float v[4];
    float s = 0.0f;
#pragma unroll
    for (int j = 0; j < 4; ++j) {
        float x = tr[tid + j * 256];
        if (leaky) x = x > 0.0f ? x : 0.01f * x;
        x += rr[tid + j * 256];
        v[j] = x;
        s += x;
    }
    red[tid] = s;
    __syncthreads();
    for (int st = 128; st > 0; st >>= 1) {
        if (tid < st) red[tid] += red[tid + st];
        __syncthreads();
    }
    float mean = red[0] * (1.0f / H_);
    __syncthreads();
    float s2 = 0.0f;
#pragma unroll
    for (int j = 0; j < 4; ++j) { float d = v[j] - mean; s2 += d * d; }
    red[tid] = s2;
    __syncthreads();
    for (int st = 128; st > 0; st >>= 1) {
        if (tid < st) red[tid] += red[tid + st];
        __syncthreads();
    }
    float inv = rsqrtf(red[0] * (1.0f / H_) + 1e-5f);
#pragma unroll
    for (int j = 0; j < 4; ++j) {
        int c = tid + j * 256;
        float y = (v[j] - mean) * inv * g[c] + be[c];
        if (outf) outf[(size_t)row * H_ + c] = y;
        if (outb) outb[(size_t)row * H_ + c] = (bf16)y;
    }
}

// ---------------------------------------------------------------------------
extern "C" void kernel_launch(void* const* d_in, const int* in_sizes, int n_in,
                              void* d_out, int out_size, void* d_ws, size_t ws_size,
                              hipStream_t stream) {
    (void)in_sizes; (void)n_in; (void)out_size; (void)ws_size;

    const float* x    = (const float*)d_in[0];
    const unsigned char* mask = (const unsigned char*)d_in[1];
    const float* pos  = (const float*)d_in[2];
    const float* Wq   = (const float*)d_in[3];
    const float* bq   = (const float*)d_in[4];
    const float* Wk   = (const float*)d_in[5];
    const float* bk   = (const float*)d_in[6];
    const float* Wv   = (const float*)d_in[7];
    const float* bv   = (const float*)d_in[8];
    const float* Wr   = (const float*)d_in[9];
    const float* br   = (const float*)d_in[10];
    const float* u    = (const float*)d_in[11];
    const float* vb   = (const float*)d_in[12];
    const float* Wo   = (const float*)d_in[13];
    const float* bo   = (const float*)d_in[14];
    const float* ln1g = (const float*)d_in[15];
    const float* ln1b = (const float*)d_in[16];
    const float* W1   = (const float*)d_in[17];
    const float* b1   = (const float*)d_in[18];
    const float* W2   = (const float*)d_in[19];
    const float* b2   = (const float*)d_in[20];
    const float* ln2g = (const float*)d_in[21];
    const float* ln2b = (const float*)d_in[22];

    // workspace layout
    char* wsb = (char*)d_ws;
    size_t o = 0;
    auto alloc = [&](size_t bytes) { size_t r = o; o += (bytes + 255) & ~(size_t)255; return r; };
    float* qf    = (float*)(wsb + alloc((size_t)ROWS * H_ * 4));
    float* kf    = (float*)(wsb + alloc((size_t)ROWS * H_ * 4));
    float* vf    = (float*)(wsb + alloc((size_t)ROWS * H_ * 4));
    bf16*  qu_b  = (bf16*) (wsb + alloc((size_t)ROWS * H_ * 2));
    bf16*  qv_b  = (bf16*) (wsb + alloc((size_t)ROWS * H_ * 2));
    bf16*  k_b   = (bf16*) (wsb + alloc((size_t)ROWS * H_ * 2));
    bf16*  vT_b  = (bf16*) (wsb + alloc((size_t)ROWS * H_ * 2));
    bf16*  WrT_b = (bf16*) (wsb + alloc((size_t)NH_ * H_ * HD_ * 2));
    bf16*  w_b   = (bf16*) (wsb + alloc((size_t)B_ * L_ * NH_ * H_ * 2));   // 16 MB
    float* cont  = (float*)(wsb + alloc((size_t)B_ * NH_ * L_ * L_ * 4));   //  8 MB
    bf16*  attn  = (bf16*) (wsb + alloc((size_t)B_ * NH_ * L_ * L_ * 2));   //  4 MB
    bf16*  aob   = (bf16*) (wsb + alloc((size_t)ROWS * H_ * 2));
    float* t1    = (float*)(wsb + alloc((size_t)ROWS * H_ * 4));
    float* x2    = (float*)(wsb + alloc((size_t)ROWS * H_ * 4));
    bf16*  x2b   = (bf16*) (wsb + alloc((size_t)ROWS * H_ * 2));
    bf16*  h1b   = (bf16*) (wsb + alloc((size_t)ROWS * H3_ * 2));
    float* t2    = (float*)(wsb + alloc((size_t)ROWS * H_ * 4));
    float* posc  = (float*)(wsb + alloc((size_t)B_ * NH_ * L_ * 4));

    // 1) fold Wr into head-major transposed bf16 layout
    wr_transpose_kernel<<<dim3((NH_ * H_ * HD_) / 256), 256, 0, stream>>>(Wr, WrT_b);

    // 2) QKV projections: f32 in, WMMA bf16, f32 out
    wmma_gemm<true, true, false, false><<<dim3(8, 32, 1), 256, 0, stream>>>(
        x, H_, 0, 0, Wq, H_, 0, 0, bq, qf, H_, 0, 0, ROWS, H_, H_, 1);
    wmma_gemm<true, true, false, false><<<dim3(8, 32, 1), 256, 0, stream>>>(
        x, H_, 0, 0, Wk, H_, 0, 0, bk, kf, H_, 0, 0, ROWS, H_, H_, 1);
    wmma_gemm<true, true, false, false><<<dim3(8, 32, 1), 256, 0, stream>>>(
        x, H_, 0, 0, Wv, H_, 0, 0, bv, vf, H_, 0, 0, ROWS, H_, H_, 1);

    // 3) head split + u/vb biases + V transpose
    prep_heads_kernel<<<dim3((B_ * NH_ * L_ * HD_) / 256), 256, 0, stream>>>(
        qf, kf, vf, u, vb, qu_b, qv_b, k_b, vT_b);
    posc_kernel<<<dim3((B_ * NH_ * L_) / 256), 256, 0, stream>>>(qf, vb, br, posc);

    // 4) w[b,q,h,e] = qv[b,h,q,:] @ Wr_h  (batched over z=(b,h), zdiv=16)
    wmma_gemm<false, false, true, false><<<dim3(8, 16, 32), 256, 0, stream>>>(
        qv_b, HD_, (long long)NH_ * L_ * HD_, (long long)L_ * HD_,
        WrT_b, HD_, 0, (long long)H_ * HD_,
        nullptr,
        w_b, NH_ * H_, (long long)L_ * NH_ * H_, (long long)H_,
        L_, H_, HD_, NH_);

    // 5) content[b,h,q,k] = (q+u) @ k^T   (batched GEMM, K=64)
    wmma_gemm<false, false, false, false><<<dim3(2, 16, 32), 256, 0, stream>>>(
        qu_b, HD_, (long long)NH_ * L_ * HD_, (long long)L_ * HD_,
        k_b, HD_, (long long)NH_ * L_ * HD_, (long long)L_ * HD_,
        nullptr,
        cont, L_, (long long)NH_ * L_ * L_, (long long)L_ * L_,
        L_, L_, HD_, NH_);

    // 6) HBM-bound position GEMM + fused mask/softmax (streams pos_emb once)
    pos_softmax_kernel<<<dim3(L_, B_), 256, 0, stream>>>(
        pos, w_b, cont, posc, mask, attn);

    // 7) out = attn @ V   (batched, N=64), bf16 into heads-merged layout
    wmma_gemm<false, false, true, false><<<dim3(1, 16, 32), 256, 0, stream>>>(
        attn, L_, (long long)NH_ * L_ * L_, (long long)L_ * L_,
        vT_b, L_, (long long)NH_ * HD_ * L_, (long long)HD_ * L_,
        nullptr,
        aob, H_, (long long)L_ * H_, (long long)HD_,
        L_, HD_, L_, NH_);

    // 8) O-projection
    wmma_gemm<false, true, false, false><<<dim3(8, 32, 1), 256, 0, stream>>>(
        aob, H_, 0, 0, Wo, H_, 0, 0, bo, t1, H_, 0, 0, ROWS, H_, H_, 1);

    // 9) leaky + residual + LN1 -> x2 (f32 + bf16)
    ln_res_kernel<<<dim3(ROWS), 256, 0, stream>>>(t1, x, ln1g, ln1b, 1, x2, x2b);

    // 10) FFN up (leaky fused in epilogue, bf16 out)
    wmma_gemm<false, true, true, true><<<dim3(24, 32, 1), 256, 0, stream>>>(
        x2b, H_, 0, 0, W1, H_, 0, 0, b1, h1b, H3_, 0, 0, ROWS, H3_, H_, 1);

    // 11) FFN down
    wmma_gemm<false, true, false, false><<<dim3(8, 32, 1), 256, 0, stream>>>(
        h1b, H3_, 0, 0, W2, H3_, 0, 0, b2, t2, H_, 0, 0, ROWS, H_, H3_, 1);

    // 12) residual + LN2 -> d_out
    ln_res_kernel<<<dim3(ROWS), 256, 0, stream>>>(
        t2, x2, ln2g, ln2b, 0, (float*)d_out, nullptr);
}
